// Attention_73813307949177
// MI455X (gfx1250) — matrix operand order repaired and checked by reading, not compiled, not run on hardware
//
#include <hip/hip_runtime.h>
#include <hip/hip_bf16.h>

typedef __bf16 bf16;
typedef __attribute__((ext_vector_type(16))) __bf16 bf16x16;
typedef __attribute__((ext_vector_type(8)))  __bf16 bf16x8;
typedef __attribute__((ext_vector_type(8)))  float   f32x8;
typedef int v4i __attribute__((vector_size(16)));  // matches builtin's V4i

#define DIM 1024
#define NQS 4096
#define NKS 4096

// ---- CDNA5 async global->LDS path (guarded; sync fallback keeps compile safe)
#if __has_builtin(__builtin_amdgcn_global_load_async_to_lds_b128)
#define HAS_ASYNC 1
#else
#define HAS_ASYNC 0
#endif

#if HAS_ASYNC
#if __has_builtin(__builtin_amdgcn_s_wait_asynccnt)
#define WAIT_ASYNC() __builtin_amdgcn_s_wait_asynccnt(0)
#else
#define WAIT_ASYNC() asm volatile("s_wait_asynccnt 0x0" ::: "memory")
#endif
#else
#define WAIT_ASYNC()
#endif

#define GAS __attribute__((address_space(1)))
#define LAS __attribute__((address_space(3)))

// ---------------- fp32 -> bf16 convert (vectorized x8) ----------------
typedef __attribute__((ext_vector_type(8))) float f32v8;
__global__ void cvt_f32_bf16_v8(const float* __restrict__ in, bf16* __restrict__ out,
                                int n8) {
  int i = blockIdx.x * blockDim.x + threadIdx.x;
  if (i < n8) {
    f32v8 v = *((const f32v8*)in + i);
    bf16x8 o;
#pragma unroll
    for (int j = 0; j < 8; ++j) o[j] = (bf16)v[j];
    *((bf16x8*)out + i) = o;
  }
}

// ---------------- bf16 tiled transpose: out[c][r] = in[r][c] ----------------
__global__ void transpose_bf16(const bf16* __restrict__ in, bf16* __restrict__ out,
                               int rows, int cols) {
  __shared__ bf16 tile[32][33];
  int r0 = blockIdx.y * 32, c0 = blockIdx.x * 32;
  int tx = threadIdx.x & 31, ty = threadIdx.x >> 5;  // 256 threads: ty = 0..7
  for (int i = ty; i < 32; i += 8)
    tile[i][tx] = in[(size_t)(r0 + i) * cols + (c0 + tx)];
  __syncthreads();
  for (int i = ty; i < 32; i += 8)
    out[(size_t)(c0 + i) * rows + (r0 + tx)] = tile[tx][i];
}

// ---------------- WMMA helper ----------------
static __device__ __forceinline__ f32x8 wmma_bf16(bf16x16 a, bf16x16 b, f32x8 c) {
  return __builtin_amdgcn_wmma_f32_16x16x32_bf16(false, a, false, b, (short)0, c,
                                                 false, false);
}

static __device__ __forceinline__ bf16x16 frag16(const bf16* p0, const bf16* p1) {
  bf16x8 lo = *(const bf16x8*)p0;
  bf16x8 hi = *(const bf16x8*)p1;
  return __builtin_shufflevector(lo, hi, 0, 1, 2, 3, 4, 5, 6, 7, 8, 9, 10, 11, 12,
                                 13, 14, 15);
}

// ---------------- GEMM-NT: C[M,N] = A[M,K] * B[N,K]^T (+bias) ----------------
#define BM 128
#define BN 128
#define KT 32
#define LDSP 40  // padded LDS row stride (elements)

// Stage a ROWSxKT bf16 tile: global -> LDS, one b128 chunk per thread per iter.
template <int ROWS>
static __device__ __forceinline__ void stage_tile(const bf16* __restrict__ g,
                                                  size_t ldg, bf16* lds, int tid) {
  constexpr int ITERS = (ROWS * KT / 8) / 256;  // exact, no bounds check
#pragma unroll
  for (int it = 0; it < ITERS; ++it) {
    int c = tid + it * 256;
    int r = c >> 2;              // KT/8 = 4 chunks per row
    int col = (c & 3) << 3;
    const bf16* gp = g + (size_t)r * ldg + col;
    bf16* lp = lds + r * LDSP + col;
#if HAS_ASYNC
    __builtin_amdgcn_global_load_async_to_lds_b128((GAS v4i*)gp, (LAS v4i*)lp, 0, 0);
#else
    *(bf16x8*)lp = *(const bf16x8*)gp;
#endif
  }
}

__global__ __launch_bounds__(256)
void gemm_nt_bf16(const bf16* __restrict__ A, const bf16* __restrict__ B,
                  const float* __restrict__ bias,
                  float* __restrict__ outF, bf16* __restrict__ outB,
                  int M, int N, int K) {
  __shared__ bf16 As[2][BM][LDSP];
  __shared__ bf16 Bs[2][BN][LDSP];

  const int tid  = threadIdx.x;
  const int lane = tid & 31;
  const int wid  = tid >> 5;      // 8 waves
  const int wm   = wid & 3;       // 4 wave rows: 32 rows each
  const int wn   = wid >> 2;      // 2 wave cols: 64 cols each
  const int blockM = blockIdx.y * BM;
  const int blockN = blockIdx.x * BN;

  const int fr = lane & 15;       // fragment row/col index
  const int fh = lane >> 4;       // fragment half select

  const f32x8 z = {0.f, 0.f, 0.f, 0.f, 0.f, 0.f, 0.f, 0.f};
  f32x8 acc[2][4];
#pragma unroll
  for (int i = 0; i < 2; ++i)
#pragma unroll
    for (int j = 0; j < 4; ++j) acc[i][j] = z;

  const bf16* Abase = A + (size_t)blockM * K;
  const bf16* Bbase = B + (size_t)blockN * K;
  const int nkt = K / KT;

  // prologue: stage tile 0 into buffer 0
  stage_tile<BM>(Abase, K, &As[0][0][0], tid);
  stage_tile<BN>(Bbase, K, &Bs[0][0][0], tid);

  for (int kt = 0; kt < nkt; ++kt) {
    const int cur = kt & 1;
    WAIT_ASYNC();
    __syncthreads();  // staged buffer `cur` visible to all waves

    if (kt + 1 < nkt) {  // issue async stage of next tile into other buffer
      stage_tile<BM>(Abase + (size_t)(kt + 1) * KT, K, &As[cur ^ 1][0][0], tid);
      stage_tile<BN>(Bbase + (size_t)(kt + 1) * KT, K, &Bs[cur ^ 1][0][0], tid);
    }

    // fragment gathers (ISA 16-bit A/B layouts: two b128 per fragment)
    bf16x16 a[2], b[4];
#pragma unroll
    for (int i = 0; i < 2; ++i) {
      const bf16* r0 = &As[cur][wm * 32 + i * 16 + fr][0];
      a[i] = frag16(r0 + fh * 8, r0 + fh * 8 + 16);
    }
#pragma unroll
    for (int j = 0; j < 4; ++j) {
      const bf16* c0 = &Bs[cur][wn * 64 + j * 16 + fr][0];
      b[j] = frag16(c0 + fh * 8, c0 + fh * 8 + 16);
    }
#pragma unroll
    for (int i = 0; i < 2; ++i)
#pragma unroll
      for (int j = 0; j < 4; ++j) acc[i][j] = wmma_bf16(a[i], b[j], acc[i][j]);
  }

  // epilogue: C frag layout — VGPR r: M = r + (lane>=16)*8, N = lane&15
  const int cm = fh * 8;
  const int cn = fr;
#pragma unroll
  for (int fi = 0; fi < 2; ++fi)
#pragma unroll
    for (int fj = 0; fj < 4; ++fj)
#pragma unroll
      for (int r = 0; r < 8; ++r) {
        size_t gm = (size_t)blockM + wm * 32 + fi * 16 + cm + r;
        size_t gn = (size_t)blockN + wn * 64 + fj * 16 + cn;
        float v = acc[fi][fj][r];
        if (bias) v += bias[gn];
        if (outF) outF[gm * N + gn] = v;
        if (outB) outB[gm * N + gn] = (bf16)v;
      }
}

// ---------------- fused double softmax over one row of score ----------------
__global__ __launch_bounds__(256)
void softmax2_rows(float* __restrict__ score, const int* __restrict__ mask,
                   const float* __restrict__ wei, bf16* __restrict__ s2B) {
  __shared__ float red[256];
  const int row = blockIdx.x;
  const int t = threadIdx.x;
  const size_t base = (size_t)row * NKS;
  const int VPT = NKS / 256;  // 16

  float x[16], b[16], w[16];
#pragma unroll
  for (int i = 0; i < VPT; ++i) {
    int c = i * 256 + t;
    b[i] = mask[base + c] ? 0.0f : -1.0e16f;
    w[i] = wei[base + c];
    x[i] = score[base + c] + b[i];
  }

  float lm = x[0];
#pragma unroll
  for (int i = 1; i < VPT; ++i) lm = fmaxf(lm, x[i]);
  red[t] = lm; __syncthreads();
  for (int s = 128; s > 0; s >>= 1) { if (t < s) red[t] = fmaxf(red[t], red[t + s]); __syncthreads(); }
  float m = red[0]; __syncthreads();
  float ls = 0.f;
#pragma unroll
  for (int i = 0; i < VPT; ++i) { x[i] = __expf(x[i] - m); ls += x[i]; }
  red[t] = ls; __syncthreads();
  for (int s = 128; s > 0; s >>= 1) { if (t < s) red[t] += red[t + s]; __syncthreads(); }
  float inv = 1.0f / red[0]; __syncthreads();

#pragma unroll
  for (int i = 0; i < VPT; ++i) x[i] = x[i] * inv * w[i] + b[i];
  lm = x[0];
#pragma unroll
  for (int i = 1; i < VPT; ++i) lm = fmaxf(lm, x[i]);
  red[t] = lm; __syncthreads();
  for (int s = 128; s > 0; s >>= 1) { if (t < s) red[t] = fmaxf(red[t], red[t + s]); __syncthreads(); }
  m = red[0]; __syncthreads();
  ls = 0.f;
#pragma unroll
  for (int i = 0; i < VPT; ++i) { x[i] = __expf(x[i] - m); ls += x[i]; }
  red[t] = ls; __syncthreads();
  for (int s = 128; s > 0; s >>= 1) { if (t < s) red[t] += red[t + s]; __syncthreads(); }
  inv = 1.0f / red[0];

#pragma unroll
  for (int i = 0; i < VPT; ++i) {
    int c = i * 256 + t;
    float v = x[i] * inv;
    score[base + c] = v;      // final score2 (fp32, already in d_out)
    s2B[base + c] = (bf16)v;  // bf16 operand for score2 @ kx
  }
}

// ---------------- host-side orchestration ----------------
extern "C" void kernel_launch(void* const* d_in, const int* in_sizes, int n_in,
                              void* d_out, int out_size, void* d_ws, size_t ws_size,
                              hipStream_t stream) {
  (void)in_sizes; (void)n_in; (void)out_size; (void)ws_size;
  const float* q    = (const float*)d_in[0];
  const float* k    = (const float*)d_in[1];
  const int*   mask = (const int*)d_in[2];
  const float* wei  = (const float*)d_in[3];
  const float* Wq   = (const float*)d_in[4];
  const float* bq   = (const float*)d_in[5];
  const float* Wk   = (const float*)d_in[6];
  const float* bk   = (const float*)d_in[7];
  const float* Wp   = (const float*)d_in[8];
  const float* bp   = (const float*)d_in[9];

  float* outF   = (float*)d_out;                      // [NQ, DIM]
  float* scoreF = (float*)d_out + (size_t)NQS * DIM;  // [NQ, NK]

  char* w = (char*)d_ws;
  size_t off = 0;
  auto walloc = [&](size_t bytes) -> void* {
    void* p = w + off;
    off += (bytes + 255) & ~(size_t)255;
    return p;
  };
  bf16* qB   = (bf16*)walloc((size_t)NQS * DIM * 2);
  bf16* kB   = (bf16*)walloc((size_t)NKS * DIM * 2);
  bf16* WqB  = (bf16*)walloc((size_t)DIM * DIM * 2);
  bf16* WkB  = (bf16*)walloc((size_t)DIM * DIM * 2);
  bf16* WpB  = (bf16*)walloc((size_t)DIM * DIM * 2);
  bf16* qxB  = (bf16*)walloc((size_t)NQS * DIM * 2);
  bf16* kxB  = (bf16*)walloc((size_t)NKS * DIM * 2);
  bf16* kxTB = (bf16*)walloc((size_t)DIM * NKS * 2);
  bf16* s2B  = (bf16*)walloc((size_t)NQS * NKS * 2);
  bf16* tmpB = (bf16*)walloc((size_t)NQS * DIM * 2);

  // 1) downconvert operands to bf16 (vectorized)
  int nq8 = NQS * DIM / 8, nk8 = NKS * DIM / 8, nw8 = DIM * DIM / 8;
  cvt_f32_bf16_v8<<<(nq8 + 255) / 256, 256, 0, stream>>>(q, qB, nq8);
  cvt_f32_bf16_v8<<<(nk8 + 255) / 256, 256, 0, stream>>>(k, kB, nk8);
  cvt_f32_bf16_v8<<<(nw8 + 255) / 256, 256, 0, stream>>>(Wq, WqB, nw8);
  cvt_f32_bf16_v8<<<(nw8 + 255) / 256, 256, 0, stream>>>(Wk, WkB, nw8);
  cvt_f32_bf16_v8<<<(nw8 + 255) / 256, 256, 0, stream>>>(Wp, WpB, nw8);

  // 2) projections: kx = k*Wk^T + bk ; qx = q*Wq^T + bq  (bf16 out)
  gemm_nt_bf16<<<dim3(DIM / BN, NKS / BM), 256, 0, stream>>>(
      kB, WkB, bk, nullptr, kxB, NKS, DIM, DIM);
  gemm_nt_bf16<<<dim3(DIM / BN, NQS / BM), 256, 0, stream>>>(
      qB, WqB, bq, nullptr, qxB, NQS, DIM, DIM);

  // kx^T so score2 @ kx stays NT
  transpose_bf16<<<dim3(DIM / 32, NKS / 32), 256, 0, stream>>>(kxB, kxTB, NKS, DIM);

  // 3) score = qx * kx^T (fp32 into d_out score region, reused as scratch)
  gemm_nt_bf16<<<dim3(NKS / BN, NQS / BM), 256, 0, stream>>>(
      qxB, kxB, nullptr, scoreF, nullptr, NQS, NKS, DIM);

  // 4) fused double softmax (mask bias + wei reweighting)
  softmax2_rows<<<NQS, 256, 0, stream>>>(scoreF, mask, wei, s2B);

  // 5) tmp = score2 * kx = score2 * (kx^T)^T (bf16 out)
  gemm_nt_bf16<<<dim3(DIM / BN, NQS / BM), 256, 0, stream>>>(
      s2B, kxTB, nullptr, nullptr, tmpB, NQS, DIM, NKS);

  // 6) out = tmp * Wp^T + bp (fp32 into d_out)
  gemm_nt_bf16<<<dim3(DIM / BN, NQS / BM), 256, 0, stream>>>(
      tmpB, WpB, bp, outF, nullptr, NQS, DIM, DIM);
}
